// DeepSeekBlock_27908697489571
// MI455X (gfx1250) — matrix operand order repaired and checked by reading, not compiled, not run on hardware
//
#include <hip/hip_runtime.h>
#include <hip/hip_bf16.h>
#include <math.h>

#define V_  32000
#define H_  768
#define NH_ 12
#define L_  2
#define E_  8
#define F_  1536
#define HD_ 64
#define B_  2
#define S_  2048
#define T_  4096
#define CAP_ 640

typedef __attribute__((ext_vector_type(16))) _Float16 v16h;
typedef __attribute__((ext_vector_type(8)))  float    v8f;

union Frag { uint4 u[2]; v16h h; };

// LDS rows padded to 40 halves (80B): 16B-aligned b128 rows, non-pow2 bank stride.
#define PITCH_    40

// ---------------------------------------------------------------------------
// CDNA5 async global -> LDS copies (ASYNCcnt-tracked), ISA 15.18.3 ops 97/98.
// ---------------------------------------------------------------------------
__device__ __forceinline__ void async_copy_b128(unsigned lds_off, const _Float16* gaddr) {
    asm volatile("global_load_async_to_lds_b128 %0, %1, off"
                 :: "v"(lds_off), "v"(gaddr) : "memory");
}
__device__ __forceinline__ void async_copy_b64(unsigned lds_off, const _Float16* gaddr) {
    asm volatile("global_load_async_to_lds_b64 %0, %1, off"
                 :: "v"(lds_off), "v"(gaddr) : "memory");
}
template<int N>
__device__ __forceinline__ void wait_asynccnt() {
    asm volatile("s_wait_asynccnt %0" :: "i"(N) : "memory");
}

// ---------------------------------------------------------------------------
// Small GEMM (for N multiples of 64 only): block 256 thr = 8 waves -> 64x... no:
// block tile 32(M) x 64(N), one 16x16 WMMA tile per wave per 32-wide K step.
// Uniform staging: every thread issues 1 async b128 (B) + 1 async b64 (A).
// C[z] = act(alpha * A[z] x B[z]^T + bias[z]) (+resid); A:[M,K], B:[N,K] f16.
// ---------------------------------------------------------------------------
#define SAROWS_ 32
#define SBROWS_ 64
#define STROWS_ (SAROWS_ + SBROWS_)

__global__ void k_gemm(const _Float16* __restrict__ A, const _Float16* __restrict__ B,
                       const float* __restrict__ bias, const float* __restrict__ resid,
                       float* __restrict__ C32, _Float16* __restrict__ C16,
                       int M, int N, int K, int lda, int ldb, int ldc,
                       long long sA1, long long sA2, long long sB1, long long sB2,
                       long long sC1, long long sC2, long long sBias1, int zdiv,
                       float alpha, int act)
{
    __shared__ __align__(16) _Float16 smem[2 * STROWS_ * PITCH_];

    int z  = blockIdx.z;
    int z1 = z / zdiv, z2 = z % zdiv;
    A += z1 * sA1 + z2 * sA2;
    B += z1 * sB1 + z2 * sB2;
    long long co = (long long)z1 * sC1 + (long long)z2 * sC2;
    if (C32)   C32   += co;
    if (C16)   C16   += co;
    if (resid) resid += co;
    if (bias)  bias  += (long long)z1 * sBias1;

    int tid  = threadIdx.x;
    int lane = tid & 31;
    int wave = tid >> 5;
    int mblk = blockIdx.y * 32;
    int nblk = blockIdx.x * 64;
    int wm = (wave >> 2) * 16;
    int wn = (wave & 3) * 16;

    // uniform staging assignment
    int chB  = tid & 3;        // 16B chunk in 64B row
    int rB   = tid >> 2;       // 0..63
    int chA  = tid & 7;        // 8B chunk in 64B row
    int rA   = tid >> 3;       // 0..31

    unsigned sbase = (unsigned)(size_t)&smem[0];      // flat addr[31:0] == LDS offset
    const unsigned bufBytes = STROWS_ * PITCH_ * 2;
    const unsigned bOffB    = SAROWS_ * PITCH_ * 2;

    auto stage = [&](int k0, int buf) {
        unsigned base = sbase + (unsigned)buf * bufBytes;
        async_copy_b128(base + bOffB + (unsigned)(rB * PITCH_ + chB * 8) * 2,
                        B + (size_t)(nblk + rB) * ldb + k0 + chB * 8);
        async_copy_b64(base + (unsigned)(rA * PITCH_ + chA * 4) * 2,
                       A + (size_t)(mblk + rA) * lda + k0 + chA * 4);
    };

    int fr  = lane & 15;
    int kbA = (lane < 16) ? 0 : 8;
    int kbB = (lane < 16) ? 0 : 8;

    v8f acc = {};
    int nsteps = K >> 5;

    stage(0, 0);
    for (int s = 0; s < nsteps; ++s) {
        int buf = s & 1;
        if ((s + 1) < nsteps) { stage((s + 1) << 5, buf ^ 1); wait_asynccnt<2>(); }
        else                  { wait_asynccnt<0>(); }
        __syncthreads();

        const _Float16* Ab = smem + buf * (STROWS_ * PITCH_);
        const _Float16* Bb = Ab + SAROWS_ * PITCH_;
        Frag af, bf;
        const _Float16* ap = Ab + (wm + fr) * PITCH_ + kbA;
        af.u[0] = *(const uint4*)(ap);
        af.u[1] = *(const uint4*)(ap + 16);
        const _Float16* bp = Bb + (wn + fr) * PITCH_ + kbB;
        bf.u[0] = *(const uint4*)(bp);
        bf.u[1] = *(const uint4*)(bp + 16);
        acc = __builtin_amdgcn_wmma_f32_16x16x32_f16(
            false, af.h, false, bf.h, (short)0, acc, false, false);

        __syncthreads();
    }

    int mbase = mblk + wm + ((lane < 16) ? 0 : 8);
    int nn = nblk + wn + fr;
    float bv = bias ? bias[nn] : 0.0f;
    #pragma unroll
    for (int r = 0; r < 8; r++) {
        int mr = mbase + r;
        float v = acc[r] * alpha + bv;
        if (act == 1) v = 0.5f * v * (1.0f + erff(v * 0.70710678118654752f));
        size_t off = (size_t)mr * ldc + nn;
        if (resid) v += resid[off];
        if (C32) C32[off] = v;
        if (C16) C16[off] = (_Float16)v;
    }
}

// ---------------------------------------------------------------------------
// Big GEMM (M%64==0, N%256==0): block tile 64(M) x 256(N), waves 2x4, each wave
// a 32x64 register tile = 2 A-frags x 4 B-frags -> 8 WMMAs per 32-wide K step.
// Uniform staging: every thread issues 5 async b128 per step (1 A + 4 B).
// Double-buffered LDS: 2 * (64+256) rows * 80B = 51.2 KB.
// ---------------------------------------------------------------------------
#define GAROWS_ 64
#define GBROWS_ 256
#define GTROWS_ (GAROWS_ + GBROWS_)

__global__ void k_gemm_big(const _Float16* __restrict__ A, const _Float16* __restrict__ B,
                           const float* __restrict__ bias, const float* __restrict__ resid,
                           float* __restrict__ C32, _Float16* __restrict__ C16,
                           int M, int N, int K, int lda, int ldb, int ldc,
                           long long sA1, long long sA2, long long sB1, long long sB2,
                           long long sC1, long long sC2, long long sBias1, int zdiv,
                           float alpha, int act)
{
    __shared__ __align__(16) _Float16 smem[2 * GTROWS_ * PITCH_];

    int z  = blockIdx.z;
    int z1 = z / zdiv, z2 = z % zdiv;
    A += z1 * sA1 + z2 * sA2;
    B += z1 * sB1 + z2 * sB2;
    long long co = (long long)z1 * sC1 + (long long)z2 * sC2;
    if (C32)   C32   += co;
    if (C16)   C16   += co;
    if (resid) resid += co;
    if (bias)  bias  += (long long)z1 * sBias1;

    int tid  = threadIdx.x;
    int lane = tid & 31;
    int wave = tid >> 5;
    int mblk = blockIdx.y * 64;
    int nblk = blockIdx.x * 256;
    int wm = (wave >> 2) * 32;            // wave row origin in A tile (0/32)
    int wn = (wave & 3) * 64;             // wave row origin in B tile (0..192)

    int ch = tid & 3;                     // 16B chunk
    int rr = tid >> 2;                    // 0..63

    unsigned sbase = (unsigned)(size_t)&smem[0];
    const unsigned bufBytes = GTROWS_ * PITCH_ * 2;
    const unsigned bOffB    = GAROWS_ * PITCH_ * 2;

    auto stage = [&](int k0, int buf) {
        unsigned base = sbase + (unsigned)buf * bufBytes;
        async_copy_b128(base + (unsigned)(rr * PITCH_ + ch * 8) * 2,
                        A + (size_t)(mblk + rr) * lda + k0 + ch * 8);
        #pragma unroll
        for (int i = 0; i < 4; i++) {
            int br = rr + 64 * i;
            async_copy_b128(base + bOffB + (unsigned)(br * PITCH_ + ch * 8) * 2,
                            B + (size_t)(nblk + br) * ldb + k0 + ch * 8);
        }
    };

    int fr  = lane & 15;
    int kbA = (lane < 16) ? 0 : 8;
    int kbB = (lane < 16) ? 0 : 8;

    v8f acc[2][4] = {};
    int nsteps = K >> 5;

    stage(0, 0);
    for (int s = 0; s < nsteps; ++s) {
        int buf = s & 1;
        if ((s + 1) < nsteps) { stage((s + 1) << 5, buf ^ 1); wait_asynccnt<5>(); }
        else                  { wait_asynccnt<0>(); }
        __syncthreads();

        const _Float16* Ab = smem + buf * (GTROWS_ * PITCH_);
        const _Float16* Bb = Ab + GAROWS_ * PITCH_;
        Frag af[2], bf[4];
        #pragma unroll
        for (int i = 0; i < 2; i++) {
            const _Float16* ap = Ab + (wm + i * 16 + fr) * PITCH_ + kbA;
            af[i].u[0] = *(const uint4*)(ap);
            af[i].u[1] = *(const uint4*)(ap + 16);
        }
        #pragma unroll
        for (int j = 0; j < 4; j++) {
            const _Float16* bp = Bb + (wn + j * 16 + fr) * PITCH_ + kbB;
            bf[j].u[0] = *(const uint4*)(bp);
            bf[j].u[1] = *(const uint4*)(bp + 16);
        }
        #pragma unroll
        for (int i = 0; i < 2; i++)
            #pragma unroll
            for (int j = 0; j < 4; j++)
                acc[i][j] = __builtin_amdgcn_wmma_f32_16x16x32_f16(
                    false, af[i].h, false, bf[j].h, (short)0, acc[i][j], false, false);

        __syncthreads();
    }

    int hi = (lane < 16) ? 0 : 8;
    #pragma unroll
    for (int i = 0; i < 2; i++) {
        #pragma unroll
        for (int j = 0; j < 4; j++) {
            int m0 = mblk + wm + i * 16 + hi;
            int nn = nblk + wn + j * 16 + fr;
            float bv = bias ? bias[nn] : 0.0f;
            #pragma unroll
            for (int r = 0; r < 8; r++) {
                int mr = m0 + r;
                float v = acc[i][j][r] * alpha + bv;
                if (act == 1) v = 0.5f * v * (1.0f + erff(v * 0.70710678118654752f));
                size_t off = (size_t)mr * ldc + nn;
                if (resid) v += resid[off];
                if (C32) C32[off] = v;
                if (C16) C16[off] = (_Float16)v;
            }
        }
    }
}

// ---------------------------------------------------------------------------
// Helper kernels
// ---------------------------------------------------------------------------
__global__ void k_convert(const float* __restrict__ in, _Float16* __restrict__ out, long long n) {
    long long i = (long long)blockIdx.x * blockDim.x + threadIdx.x;
    if (i < n) out[i] = (_Float16)in[i];
}

// [z][R][C] f32 -> [z][C][R] f16
__global__ void k_tconvert(const float* __restrict__ in, _Float16* __restrict__ out, int R, int C) {
    long long z = blockIdx.z;
    in  += z * (long long)R * C;
    out += z * (long long)R * C;
    int idx = blockIdx.x * blockDim.x + threadIdx.x;
    if (idx < R * C) {
        int r = idx / C, c = idx % C;
        out[(long long)c * R + r] = (_Float16)in[idx];
    }
}

__global__ void k_embed(const int* __restrict__ ids, const float* __restrict__ emb,
                        float* __restrict__ h) {
    int t = blockIdx.x;
    long long id = ids[t];
    for (int i = threadIdx.x; i < H_; i += blockDim.x)
        h[(long long)t * H_ + i] = emb[id * H_ + i];
}

__global__ void k_ln(const float* __restrict__ x, const float* __restrict__ g,
                     const float* __restrict__ b, float* __restrict__ y32,
                     _Float16* __restrict__ y16) {
    int t = blockIdx.x;
    const float* xr = x + (long long)t * H_;
    __shared__ float s1[256], s2[256];
    float a = 0.f, q = 0.f;
    for (int i = threadIdx.x; i < H_; i += 256) { float v = xr[i]; a += v; q += v * v; }
    s1[threadIdx.x] = a; s2[threadIdx.x] = q; __syncthreads();
    for (int o = 128; o > 0; o >>= 1) {
        if (threadIdx.x < o) { s1[threadIdx.x] += s1[threadIdx.x + o]; s2[threadIdx.x] += s2[threadIdx.x + o]; }
        __syncthreads();
    }
    float m = s1[0] / H_;
    float var = s2[0] / H_ - m * m;
    float rs = rsqrtf(var + 1e-5f);
    for (int i = threadIdx.x; i < H_; i += 256) {
        float v = (xr[i] - m) * rs * g[i] + b[i];
        if (y32) y32[(long long)t * H_ + i] = v;
        if (y16) y16[(long long)t * H_ + i] = (_Float16)v;
    }
}

// v16 [B*S, HD] -> vT16 [B, HD, S]
__global__ void k_vtrans(const _Float16* __restrict__ v, _Float16* __restrict__ vt) {
    int bs = blockIdx.x; int b = bs / S_, s = bs % S_;
    int d = threadIdx.x; // blockDim = HD_
    vt[((long long)b * HD_ + d) * S_ + s] = v[(long long)bs * HD_ + d];
}

// softmax over keys with mask; rows = B*NH*S, cols = S
__global__ void k_attn_softmax(const float* __restrict__ sc, const float* __restrict__ mask,
                               _Float16* __restrict__ pr) {
    long long row = blockIdx.x;
    int b = (int)(row / ((long long)NH_ * S_));
    const float* srow = sc + row * S_;
    const float* mrow = mask + (long long)b * S_;
    __shared__ float red[256];
    float ev[S_ / 256];
    float mx = -3.4e38f;
    #pragma unroll
    for (int j = 0; j < S_ / 256; j++) {
        int k = threadIdx.x + j * 256;
        float v = srow[k] + (1.0f - mrow[k]) * (-3.4028234663852886e38f);
        ev[j] = v; mx = fmaxf(mx, v);
    }
    red[threadIdx.x] = mx; __syncthreads();
    for (int o = 128; o > 0; o >>= 1) {
        if (threadIdx.x < o) red[threadIdx.x] = fmaxf(red[threadIdx.x], red[threadIdx.x + o]);
        __syncthreads();
    }
    mx = red[0]; __syncthreads();
    float sm = 0.f;
    #pragma unroll
    for (int j = 0; j < S_ / 256; j++) { ev[j] = expf(ev[j] - mx); sm += ev[j]; }
    red[threadIdx.x] = sm; __syncthreads();
    for (int o = 128; o > 0; o >>= 1) {
        if (threadIdx.x < o) red[threadIdx.x] += red[threadIdx.x + o];
        __syncthreads();
    }
    float inv = 1.0f / red[0];
    #pragma unroll
    for (int j = 0; j < S_ / 256; j++) {
        int k = threadIdx.x + j * 256;
        pr[row * S_ + k] = (_Float16)(ev[j] * inv);
    }
}

// gate scores = softmax(x @ gw^T) per token; gw [E,H]
__global__ void k_gate(const float* __restrict__ x, const float* __restrict__ gw,
                       float* __restrict__ out) {
    int t = blockIdx.x;
    __shared__ float red[E_][256];
    float acc[E_] = {};
    const float* xr = x + (long long)t * H_;
    for (int i = threadIdx.x; i < H_; i += 256) {
        float xv = xr[i];
        #pragma unroll
        for (int e = 0; e < E_; e++) acc[e] += xv * gw[e * H_ + i];
    }
    for (int e = 0; e < E_; e++) red[e][threadIdx.x] = acc[e];
    __syncthreads();
    for (int o = 128; o > 0; o >>= 1) {
        if (threadIdx.x < o)
            for (int e = 0; e < E_; e++) red[e][threadIdx.x] += red[e][threadIdx.x + o];
        __syncthreads();
    }
    if (threadIdx.x == 0) {
        float mx = -3.4e38f; float v[E_];
        for (int e = 0; e < E_; e++) { v[e] = red[e][0]; mx = fmaxf(mx, v[e]); }
        float s = 0.f;
        for (int e = 0; e < E_; e++) { v[e] = expf(v[e] - mx); s += v[e]; }
        for (int e = 0; e < E_; e++) out[(long long)t * E_ + e] = v[e] / s;
    }
}

// exact top-cap per expert via iterative argmax; one block per expert
__global__ void k_route(const float* __restrict__ gs, int* __restrict__ top_i,
                        float* __restrict__ top_s, float* __restrict__ load8) {
    int e = blockIdx.x;
    __shared__ float vals[T_];
    __shared__ float rv[256];
    __shared__ int   ri[256];
    for (int t = threadIdx.x; t < T_; t += 256) vals[t] = gs[(long long)t * E_ + e];
    __syncthreads();
    float sum = 0.f;
    for (int c = 0; c < CAP_; c++) {
        float bv = -3.4e38f; int bi = 0x7fffffff;
        for (int t = threadIdx.x; t < T_; t += 256) {
            float v = vals[t];
            if (v > bv) { bv = v; bi = t; }
        }
        rv[threadIdx.x] = bv; ri[threadIdx.x] = bi; __syncthreads();
        for (int o = 128; o > 0; o >>= 1) {
            if (threadIdx.x < o) {
                float ov = rv[threadIdx.x + o]; int oi = ri[threadIdx.x + o];
                if (ov > rv[threadIdx.x] || (ov == rv[threadIdx.x] && oi < ri[threadIdx.x])) {
                    rv[threadIdx.x] = ov; ri[threadIdx.x] = oi;
                }
            }
            __syncthreads();
        }
        if (threadIdx.x == 0) {
            top_i[e * CAP_ + c] = ri[0];
            top_s[e * CAP_ + c] = rv[0];
            vals[ri[0]] = -3.4e38f;
            sum += rv[0];
        }
        __syncthreads();
    }
    if (threadIdx.x == 0) load8[e] = sum / (float)T_;
}

__global__ void k_lb(const float* __restrict__ load8, float* __restrict__ out_last) {
    if (threadIdx.x == 0 && blockIdx.x == 0) {
        float m = 0.f;
        for (int e = 0; e < E_; e++) m += load8[e];
        m /= E_;
        float v = 0.f;
        for (int e = 0; e < E_; e++) { float d = load8[e] - m; v += d * d; }
        v /= (E_ - 1);                       // ddof=1
        atomicAdd(out_last, v * (float)E_);
    }
}

__global__ void k_gather(const int* __restrict__ top_i, const _Float16* __restrict__ src,
                         _Float16* __restrict__ dst) {
    int g = blockIdx.x;                      // e*CAP_+c
    long long t = top_i[g];
    for (int i = threadIdx.x; i < H_; i += 256)
        dst[(long long)g * H_ + i] = src[t * H_ + i];
}

__global__ void k_scatter(const int* __restrict__ top_i, const float* __restrict__ top_s,
                          const float* __restrict__ out_e, float* __restrict__ h) {
    int g = blockIdx.x;
    long long t = top_i[g];
    float w = top_s[g];
    for (int i = threadIdx.x; i < H_; i += 256)
        atomicAdd(&h[t * H_ + i], w * out_e[(long long)g * H_ + i]);
}

__global__ void k_zero1(float* p) {
    if (threadIdx.x == 0 && blockIdx.x == 0) p[0] = 0.0f;
}

// ---------------------------------------------------------------------------
// Host orchestration
// ---------------------------------------------------------------------------
static void gemm(hipStream_t st, const _Float16* A, const _Float16* B,
                 const float* bias, const float* resid, float* C32, _Float16* C16,
                 int M, int N, int K, int lda, int ldb, int ldc,
                 long long sA1, long long sA2, long long sB1, long long sB2,
                 long long sC1, long long sC2, long long sBias1, int zdiv, int Z,
                 float alpha, int act)
{
    if ((N % 256) == 0 && (M % 64) == 0) {
        dim3 grid(N / 256, M / 64, Z);
        k_gemm_big<<<grid, 256, 0, st>>>(A, B, bias, resid, C32, C16, M, N, K, lda, ldb, ldc,
                                         sA1, sA2, sB1, sB2, sC1, sC2, sBias1, zdiv, alpha, act);
    } else {
        dim3 grid(N / 64, M / 32, Z);
        k_gemm<<<grid, 256, 0, st>>>(A, B, bias, resid, C32, C16, M, N, K, lda, ldb, ldc,
                                     sA1, sA2, sB1, sB2, sC1, sC2, sBias1, zdiv, alpha, act);
    }
}

extern "C" void kernel_launch(void* const* d_in, const int* in_sizes, int n_in,
                              void* d_out, int out_size, void* d_ws, size_t ws_size,
                              hipStream_t stream) {
    const int*   ids    = (const int*)d_in[0];
    const float* mask   = (const float*)d_in[1];
    const float* embW   = (const float*)d_in[2];
    const float* ln1g   = (const float*)d_in[3];
    const float* ln1b   = (const float*)d_in[4];
    const float* qW     = (const float*)d_in[5];
    const float* kW     = (const float*)d_in[6];
    const float* vW     = (const float*)d_in[7];
    const float* oW     = (const float*)d_in[8];
    const float* ln2g   = (const float*)d_in[9];
    const float* ln2b   = (const float*)d_in[10];
    const float* gateW  = (const float*)d_in[11];
    const float* w1     = (const float*)d_in[12];
    const float* b1     = (const float*)d_in[13];
    const float* w2     = (const float*)d_in[14];
    const float* b2     = (const float*)d_in[15];
    const float* lnfg   = (const float*)d_in[16];
    const float* lnfb   = (const float*)d_in[17];
    const float* headW  = (const float*)d_in[18];

    float* out   = (float*)d_out;
    float* outLB = out + (out_size - 1);

    // --- workspace carve (bump allocator, 256B aligned) ---
    char* p = (char*)d_ws;
    auto alloc = [&](size_t bytes) -> char* {
        char* r = p; p += (bytes + 255) & ~(size_t)255; return r;
    };
    float*    h       = (float*)   alloc(sizeof(float)    * (size_t)T_ * H_);
    float*    lnf32   = (float*)   alloc(sizeof(float)    * (size_t)T_ * H_);
    _Float16* ln16    = (_Float16*)alloc(sizeof(_Float16) * (size_t)T_ * H_);
    _Float16* q16     = (_Float16*)alloc(sizeof(_Float16) * (size_t)T_ * H_);
    _Float16* k16     = (_Float16*)alloc(sizeof(_Float16) * (size_t)T_ * HD_);
    _Float16* v16     = (_Float16*)alloc(sizeof(_Float16) * (size_t)T_ * HD_);
    _Float16* vT16    = (_Float16*)alloc(sizeof(_Float16) * (size_t)B_ * HD_ * S_);
    float*    scores  = (float*)   alloc(sizeof(float)    * (size_t)B_ * NH_ * S_ * S_);
    _Float16* probs16 = (_Float16*)alloc(sizeof(_Float16) * (size_t)B_ * NH_ * S_ * S_);
    _Float16* attn16  = (_Float16*)alloc(sizeof(_Float16) * (size_t)T_ * H_);
    float*    gate_s  = (float*)   alloc(sizeof(float)    * (size_t)T_ * E_);
    int*      top_i   = (int*)     alloc(sizeof(int)      * (size_t)E_ * CAP_);
    float*    top_s   = (float*)   alloc(sizeof(float)    * (size_t)E_ * CAP_);
    float*    load8   = (float*)   alloc(sizeof(float)    * E_);
    _Float16* xin16   = (_Float16*)alloc(sizeof(_Float16) * (size_t)E_ * CAP_ * H_);
    _Float16* h1_16   = (_Float16*)alloc(sizeof(_Float16) * (size_t)E_ * CAP_ * F_);
    float*    out_e   = (float*)   alloc(sizeof(float)    * (size_t)E_ * CAP_ * H_);
    _Float16* qW16    = (_Float16*)alloc(sizeof(_Float16) * (size_t)L_ * H_ * H_);
    _Float16* kW16    = (_Float16*)alloc(sizeof(_Float16) * (size_t)L_ * HD_ * H_);
    _Float16* vW16    = (_Float16*)alloc(sizeof(_Float16) * (size_t)L_ * HD_ * H_);
    _Float16* oW16    = (_Float16*)alloc(sizeof(_Float16) * (size_t)L_ * H_ * H_);
    _Float16* w1T16   = (_Float16*)alloc(sizeof(_Float16) * (size_t)L_ * E_ * F_ * H_);
    _Float16* w2T16   = (_Float16*)alloc(sizeof(_Float16) * (size_t)L_ * E_ * H_ * F_);
    _Float16* headW16 = (_Float16*)alloc(sizeof(_Float16) * (size_t)V_ * H_);
    (void)ws_size; (void)n_in; (void)in_sizes;

    // --- one-time f32 -> f16 weight conversion ---
    auto conv = [&](const float* src, _Float16* dst, long long n) {
        k_convert<<<dim3((unsigned)((n + 255) / 256)), 256, 0, stream>>>(src, dst, n);
    };
    conv(qW, qW16, (long long)L_ * H_ * H_);
    conv(kW, kW16, (long long)L_ * HD_ * H_);
    conv(vW, vW16, (long long)L_ * HD_ * H_);
    conv(oW, oW16, (long long)L_ * H_ * H_);
    conv(headW, headW16, (long long)V_ * H_);
    {   // w1 [L*E][H][F] -> [L*E][F][H],  w2 [L*E][F][H] -> [L*E][H][F]
        dim3 g1((H_ * F_ + 255) / 256, 1, L_ * E_);
        k_tconvert<<<g1, 256, 0, stream>>>(w1, w1T16, H_, F_);
        k_tconvert<<<g1, 256, 0, stream>>>(w2, w2T16, F_, H_);
    }

    k_zero1<<<1, 1, 0, stream>>>(outLB);
    k_embed<<<T_, 256, 0, stream>>>(ids, embW, h);

    const float scale = 0.125f; // HD^-0.5, HD=64

    for (int l = 0; l < L_; l++) {
        const _Float16* qWl  = qW16  + (long long)l * H_ * H_;
        const _Float16* kWl  = kW16  + (long long)l * HD_ * H_;
        const _Float16* vWl  = vW16  + (long long)l * HD_ * H_;
        const _Float16* oWl  = oW16  + (long long)l * H_ * H_;
        const _Float16* w1Tl = w1T16 + (long long)l * E_ * F_ * H_;
        const _Float16* w2Tl = w2T16 + (long long)l * E_ * H_ * F_;

        // LN1 -> f16
        k_ln<<<T_, 256, 0, stream>>>(h, ln1g + l * H_, ln1b + l * H_, nullptr, ln16);

        // Q/K/V projections (WMMA)
        gemm(stream, ln16, qWl, nullptr, nullptr, nullptr, q16,
             T_, H_, H_, H_, H_, H_, 0, 0, 0, 0, 0, 0, 0, 1, 1, 1.0f, 0);
        gemm(stream, ln16, kWl, nullptr, nullptr, nullptr, k16,
             T_, HD_, H_, H_, H_, HD_, 0, 0, 0, 0, 0, 0, 0, 1, 1, 1.0f, 0);
        gemm(stream, ln16, vWl, nullptr, nullptr, nullptr, v16,
             T_, HD_, H_, H_, H_, HD_, 0, 0, 0, 0, 0, 0, 0, 1, 1, 1.0f, 0);
        k_vtrans<<<B_ * S_, HD_, 0, stream>>>(v16, vT16);

        // scores[b,n] = scale * Q[b,:,n,:] @ K[b]^T   (z = b*NH+n)
        gemm(stream, q16, k16, nullptr, nullptr, scores, nullptr,
             S_, S_, HD_, H_, HD_, S_,
             (long long)S_ * H_, HD_,
             (long long)S_ * HD_, 0,
             (long long)NH_ * S_ * S_, (long long)S_ * S_,
             0, NH_, B_ * NH_, scale, 0);

        k_attn_softmax<<<B_ * NH_ * S_, 256, 0, stream>>>(scores, mask, probs16);

        // attn[b,:,n*HD..] = P[b,n] @ V[b]   (B-matrix = V^T; N=64 -> small kernel)
        gemm(stream, probs16, vT16, nullptr, nullptr, nullptr, attn16,
             S_, HD_, S_, S_, S_, H_,
             (long long)NH_ * S_ * S_, (long long)S_ * S_,
             (long long)HD_ * S_, 0,
             (long long)S_ * H_, HD_,
             0, NH_, B_ * NH_, 1.0f, 0);

        // output projection + residual into h
        gemm(stream, attn16, oWl, nullptr, h, h, nullptr,
             T_, H_, H_, H_, H_, H_, 0, 0, 0, 0, 0, 0, 0, 1, 1, 1.0f, 0);

        // LN2 -> f32 (gate) + f16 (expert GEMM input)
        k_ln<<<T_, 256, 0, stream>>>(h, ln2g + l * H_, ln2b + l * H_, lnf32, ln16);

        // MoE routing
        k_gate<<<T_, 256, 0, stream>>>(lnf32, gateW + (long long)l * E_ * H_, gate_s);
        k_route<<<E_, 256, 0, stream>>>(gate_s, top_i, top_s, load8);
        k_lb<<<1, 1, 0, stream>>>(load8, outLB);
        k_gather<<<E_ * CAP_, 256, 0, stream>>>(top_i, ln16, xin16);

        // expert FFN: h1 = gelu(xin @ w1 + b1) ; out_e = h1 @ w2 + b2
        gemm(stream, xin16, w1Tl, b1 + (long long)l * E_ * F_, nullptr, nullptr, h1_16,
             CAP_, F_, H_, H_, H_, F_,
             (long long)CAP_ * H_, 0, (long long)F_ * H_, 0,
             (long long)CAP_ * F_, 0, F_, 1, E_, 1.0f, 1);
        gemm(stream, h1_16, w2Tl, b2 + (long long)l * E_ * H_, nullptr, out_e, nullptr,
             CAP_, H_, F_, F_, F_, H_,
             (long long)CAP_ * F_, 0, (long long)H_ * F_, 0,
             (long long)CAP_ * H_, 0, H_, 1, E_, 1.0f, 0);

        // weighted scatter-add into residual h
        k_scatter<<<E_ * CAP_, 256, 0, stream>>>(top_i, top_s, out_e, h);
    }

    // final LN + vocab head -> logits
    k_ln<<<T_, 256, 0, stream>>>(h, lnfg, lnfb, nullptr, ln16);
    gemm(stream, ln16, headW16, nullptr, nullptr, out, nullptr,
         T_, V_, H_, H_, H_, V_, 0, 0, 0, 0, 0, 0, 0, 1, 1, 1.0f, 0);
}
